// Block_66391604461805
// MI455X (gfx1250) — compile-verified, matrix operand order checked by reading, not run on hardware
//
#include <hip/hip_runtime.h>

typedef _Float16 f16;
typedef __attribute__((ext_vector_type(16))) _Float16 v16h;
typedef __attribute__((ext_vector_type(8)))  _Float16 v8h;
typedef __attribute__((ext_vector_type(8)))  float    v8f;

#define DEV __device__ __forceinline__

constexpr int DIMC   = 384;
constexpr int HEADS  = 6;
constexpr int HD     = 64;
constexpr int HIDDEN = 1536;
constexpr int NSEQ   = 4096;
constexpr int NTOK   = 8192;   // B*N

DEV v8f wmma_f16(v16h a, v16h b, v8f c) {
  // D = A(16x32) x B(32x16) + C, fp32 accumulate
  return __builtin_amdgcn_wmma_f32_16x16x32_f16(false, a, false, b, (short)0, c,
                                                false, false);
}

// Async global->LDS copy, 16B per lane. Tracked by ASYNCcnt (ISA 10.x / 15.18).
DEV void async_b128(unsigned lds_byte_off, const void* gptr) {
  asm volatile("global_load_async_to_lds_b128 %0, %1, off"
               :: "v"(lds_byte_off), "v"((unsigned long long)(size_t)gptr)
               : "memory");
}
DEV void wait_async0() { asm volatile("s_wait_asynccnt 0x0" ::: "memory"); }

// Raw workgroup split-barrier WITHOUT the memory fence of __syncthreads().
// Safe here: ds reads complete before their consuming WMMA issues (dscnt
// waits), and LDS writes are covered by s_wait_asynccnt before the signal.
// Avoids the s_wait_loadcnt 0 flush that serializes pipelined global loads.
DEV void wg_barrier() {
  asm volatile("s_barrier_signal -1\n\ts_barrier_wait -1" ::: "memory");
}

// Low 32 bits of a generic pointer to __shared__ == LDS byte offset.
DEV unsigned lds_off(const void* p) { return (unsigned)(size_t)p; }

// Load a 16x32 fragment from a row-major matrix (global or LDS).
// A-layout (ISA 7.12.2): lanes 0-15 hold rows 0-15 with K={0..7,16..23},
// lanes 16-31 hold rows 0-15 with K={8..15,24..31}. The identical pattern is
// the B fragment when B's column n is contiguous (B = W^T, W row-major).
DEV v16h frag_load(const f16* p, int ld) {
  const int lane = threadIdx.x & 31;
  const int grp  = lane >> 4;
  const int r    = lane & 15;
  const f16* q = p + (size_t)r * ld + grp * 8;
  v8h lo = *(const v8h*)(q);
  v8h hi = *(const v8h*)(q + 16);
  return __builtin_shufflevector(lo, hi, 0,1,2,3,4,5,6,7,8,9,10,11,12,13,14,15);
}

// ---------------------------------------------------------------- utilities
__global__ void cvt_kernel(const float* __restrict__ in, f16* __restrict__ out, int n) {
  int i = blockIdx.x * 256 + threadIdx.x;
  if (i < n) out[i] = (f16)in[i];
}

// LN over DIMC=384, one wave per token, 8 tokens per 256-thread block.
__global__ void ln1_kernel(const float* __restrict__ x, const float* __restrict__ w,
                           const float* __restrict__ b, f16* __restrict__ y) {
  const int tok  = blockIdx.x * 8 + (threadIdx.x >> 5);
  const int lane = threadIdx.x & 31;
  const float* xr = x + (size_t)tok * DIMC;
  float s = 0.f, s2 = 0.f;
  for (int i = lane; i < DIMC; i += 32) { float v = xr[i]; s += v; s2 += v * v; }
#pragma unroll
  for (int m = 16; m >= 1; m >>= 1) { s += __shfl_xor(s, m); s2 += __shfl_xor(s2, m); }
  const float mean = s / DIMC;
  const float inv  = rsqrtf(s2 / DIMC - mean * mean + 1e-5f);
  f16* yr = y + (size_t)tok * DIMC;
  for (int i = lane; i < DIMC; i += 32)
    yr[i] = (f16)((xr[i] - mean) * inv * w[i] + b[i]);
}

// s = x + proj_out (bias already in proj_out), then LN -> f16
__global__ void ln2_kernel(const float* __restrict__ x, const float* __restrict__ p,
                           const float* __restrict__ w, const float* __restrict__ b,
                           f16* __restrict__ y) {
  const int tok  = blockIdx.x * 8 + (threadIdx.x >> 5);
  const int lane = threadIdx.x & 31;
  const float* xr = x + (size_t)tok * DIMC;
  const float* pr = p + (size_t)tok * DIMC;
  float s = 0.f, s2 = 0.f;
  for (int i = lane; i < DIMC; i += 32) { float v = xr[i] + pr[i]; s += v; s2 += v * v; }
#pragma unroll
  for (int m = 16; m >= 1; m >>= 1) { s += __shfl_xor(s, m); s2 += __shfl_xor(s2, m); }
  const float mean = s / DIMC;
  const float inv  = rsqrtf(s2 / DIMC - mean * mean + 1e-5f);
  f16* yr = y + (size_t)tok * DIMC;
  for (int i = lane; i < DIMC; i += 32)
    yr[i] = (f16)((xr[i] + pr[i] - mean) * inv * w[i] + b[i]);
}

// ----------------------------------------------- K=384 full-panel main loop
// The 64-row W panel of a block covers ALL K columns, so it is a contiguous
// 48KB region of W. Stage it into LDS once (12 async b128 copies per thread),
// one barrier, then run the fully-unrolled k-loop with zero synchronization.
DEV void gemm_mainloop_panel384(const f16* __restrict__ Xb,
                                const f16* __restrict__ W, int n0,
                                f16* Wl /* [64*384] */, v8f acc[2][4]) {
  constexpr int K = DIMC;  // 384
  const int tid = threadIdx.x;
  const unsigned lbase = lds_off(Wl);
  const f16* gW = W + (size_t)n0 * K;   // panel start (contiguous 48KB)
#pragma unroll
  for (int i = 0; i < 12; ++i) {
    const int c = tid + i * 256;        // 16B chunk index, 3072 total
    async_b128(lbase + (unsigned)c * 16, gW + (size_t)c * 8);
  }
  const f16* Xb1 = Xb + (size_t)16 * K;
  wait_async0();
  wg_barrier();
#pragma unroll
  for (int k0 = 0; k0 < K; k0 += 32) {
    const v16h a0 = frag_load(Xb + k0, K);
    const v16h a1 = frag_load(Xb1 + k0, K);
#pragma unroll
    for (int t = 0; t < 4; ++t) {
      const v16h bt = frag_load(Wl + (size_t)(16 * t) * K + k0, K);
      acc[0][t] = wmma_f16(a0, bt, acc[0][t]);
      acc[1][t] = wmma_f16(a1, bt, acc[1][t]);
    }
  }
}

// ------------------------------------- chunked main loop (large K, e.g. 1536)
// W chunk (64 x 32) staged in LDS by async copies, double buffered; A frags
// software-pipelined in VGPRs one k-step ahead.
DEV void gemm_mainloop_chunked(const f16* __restrict__ Xb, const f16* __restrict__ W,
                               int n0, int K, f16* Wl /* [2][64*32] */,
                               v8f acc[2][4]) {
  const int tid  = threadIdx.x;
  const int srow = tid >> 2, scc = tid & 3;      // 64 rows x 4 chunks of 16B
  const unsigned lbase = lds_off(Wl);
  const f16* gW = W + (size_t)(n0 + srow) * K + scc * 8;
  const f16* Xb1 = Xb + (size_t)16 * K;

#define STAGE_W(buf, k0) \
  async_b128(lbase + (unsigned)((buf) * 4096 + srow * 64 + scc * 16), gW + (k0))

  STAGE_W(0, 0);
  v16h a0 = frag_load(Xb, K);
  v16h a1 = frag_load(Xb1, K);
  wait_async0();
  wg_barrier();

  for (int k0 = 0; k0 < K; k0 += 32) {
    const int cur = (k0 >> 5) & 1;
    v16h a0n = a0, a1n = a1;
    if (k0 + 32 < K) {
      STAGE_W(cur ^ 1, k0 + 32);
      a0n = frag_load(Xb + k0 + 32, K);
      a1n = frag_load(Xb1 + k0 + 32, K);
    }
    const f16* Wc = Wl + cur * 2048;
#pragma unroll
    for (int t = 0; t < 4; ++t) {
      const v16h bt = frag_load(Wc + t * 512, 32);
      acc[0][t] = wmma_f16(a0, bt, acc[0][t]);
      acc[1][t] = wmma_f16(a1, bt, acc[1][t]);
    }
    wait_async0();
    wg_barrier();
    a0 = a0n; a1 = a1n;
  }
#undef STAGE_W
}

// ------------------------------------------------------------- QKV GEMM
// qkv = y1 @ qkv_w^T (N=1152, K=384). which(q/k/v) is uniform per block:
// gridDim.x = 18, blocks [0,6) -> q, [6,12) -> k, [12,18) -> v.
// q scaled by HD^-0.5 -> [B,H,N,64]; k -> [B,H,N,64]; v^T -> [B,H,64,N].
__global__ void qkv_gemm_kernel(const f16* __restrict__ X, const f16* __restrict__ W,
                                f16* __restrict__ qd, f16* __restrict__ kd,
                                f16* __restrict__ vtd) {
  __shared__ alignas(16) f16 Wl[64 * DIMC];
  constexpr int K = DIMC;
  const int lane = threadIdx.x & 31;
  const int grp = lane >> 4, lr = lane & 15;
  const int n0 = blockIdx.x * 64;
  const int m0 = blockIdx.y * 256 + (threadIdx.x >> 5) * 32;

  v8f acc[2][4] = {{v8f{}, v8f{}, v8f{}, v8f{}}, {v8f{}, v8f{}, v8f{}, v8f{}}};
  gemm_mainloop_panel384(X + (size_t)m0 * K, W, n0, Wl, acc);

  const int which = blockIdx.x / 6;          // uniform: 0=q, 1=k, 2=v
  const int cbase = n0 - which * DIMC;       // column within section
#pragma unroll
  for (int t = 0; t < 4; ++t) {
    const int col = cbase + 16 * t + lr;     // 0..383
    const int h = col >> 6, d = col & 63;
#pragma unroll
    for (int mi = 0; mi < 2; ++mi) {
#pragma unroll
      for (int r = 0; r < 8; ++r) {
        const int m = m0 + mi * 16 + r + 8 * grp;  // token 0..8191
        const int bb = m >> 12;                    // batch
        const int pos = m & 4095;
        const float v = acc[mi][t][r];
        const size_t bh = (size_t)(bb * HEADS + h);
        if (which == 0)      qd[(bh * NSEQ + pos) * HD + d] = (f16)(v * 0.125f);
        else if (which == 1) kd[(bh * NSEQ + pos) * HD + d] = (f16)v;
        else                 vtd[(bh * HD + d) * NSEQ + pos] = (f16)v;
      }
    }
  }
}

// ------------------------------------------------------------- generic GEMM
enum { EPI_BIAS_F32 = 0, EPI_GELU_F16 = 1, EPI_BIAS2X_F32 = 2 };

template <int EPI, bool PANEL>
__global__ void gemm_kernel(const f16* __restrict__ X, const f16* __restrict__ W,
                            const float* __restrict__ bias, void* __restrict__ outp,
                            int N, int K) {
  __shared__ alignas(16) f16 Wl[PANEL ? 64 * DIMC : 2 * 2048];
  const int lane = threadIdx.x & 31;
  const int grp = lane >> 4, lr = lane & 15;
  const int n0 = blockIdx.x * 64;
  const int m0 = blockIdx.y * 256 + (threadIdx.x >> 5) * 32;

  v8f acc[2][4] = {{v8f{}, v8f{}, v8f{}, v8f{}}, {v8f{}, v8f{}, v8f{}, v8f{}}};
  if (PANEL)
    gemm_mainloop_panel384(X + (size_t)m0 * DIMC, W, n0, Wl, acc);
  else
    gemm_mainloop_chunked(X + (size_t)m0 * K, W, n0, K, Wl, acc);

#pragma unroll
  for (int t = 0; t < 4; ++t) {
    const int n = n0 + 16 * t + lr;
    const float bn = bias[n];
#pragma unroll
    for (int mi = 0; mi < 2; ++mi) {
#pragma unroll
      for (int r = 0; r < 8; ++r) {
        const size_t m = (size_t)(m0 + mi * 16 + r + 8 * grp);
        const float v = acc[mi][t][r] + bn;
        if (EPI == EPI_BIAS_F32) {
          ((float*)outp)[m * N + n] = v;
        } else if (EPI == EPI_GELU_F16) {
          ((f16*)outp)[m * N + n] = (f16)(0.5f * v * (1.0f + erff(v * 0.70710678118f)));
        } else {
          ((float*)outp)[m * N + n] = 2.0f * v;  // x = x + x
        }
      }
    }
  }
}

// ------------------------------------------------------------- attention
// Block = 8 waves sharing one (b,h); wave w owns query rows [m0, m0+16).
// K block (32x64) and V^T block (64x32) staged in LDS with async copies,
// double buffered; all waves consume the same tiles (8x traffic reduction).
// Q pre-scaled by HD^-0.5. K: [B,H,N,64]. VT: [B,H,64,N]. O: [B*N,384] f16.
__global__ void attn_kernel(const f16* __restrict__ Q, const f16* __restrict__ Kp,
                            const f16* __restrict__ VT, f16* __restrict__ O) {
  __shared__ alignas(16) f16 Kl[2][32 * 64];
  __shared__ alignas(16) f16 Vl[2][64 * 32];
  __shared__ alignas(16) f16 Pl[8][16 * 32];
  const int tid = threadIdx.x;
  const int wave = tid >> 5, lane = tid & 31;
  const int grp = lane >> 4, lr = lane & 15;
  constexpr int TPB = NSEQ / 128;    // 32 row-tiles-of-128 per (b,h)
  const int bh = blockIdx.x / TPB;
  const int m0 = (blockIdx.x % TPB) * 128 + wave * 16;
  const f16* Qb = Q + ((size_t)bh * NSEQ + m0) * HD;
  const f16* Kb = Kp + (size_t)bh * NSEQ * HD;
  const f16* Vb = VT + (size_t)bh * HD * NSEQ;
  f16* pt = Pl[wave];

  // staging coordinates: K tile = 32 rows x 8 16B-chunks, V tile = 64 x 4
  const int krow = tid >> 3, kcc = tid & 7;
  const int vrow = tid >> 2, vcc = tid & 3;
  const unsigned klb = lds_off(&Kl[0][0]);
  const unsigned vlb = lds_off(&Vl[0][0]);
  const f16* gK = Kb + (size_t)krow * HD + kcc * 8;
  const f16* gV = Vb + (size_t)vrow * NSEQ + vcc * 8;

#define STAGE_KV(buf, n0)                                                  \
  do {                                                                     \
    async_b128(klb + (unsigned)((buf) * 4096 + krow * 128 + kcc * 16),     \
               gK + (size_t)(n0) * HD);                                    \
    async_b128(vlb + (unsigned)((buf) * 4096 + vrow * 64 + vcc * 16),      \
               gV + (n0));                                                 \
  } while (0)

  STAGE_KV(0, 0);
  const v16h qa0 = frag_load(Qb + 0, HD);
  const v16h qa1 = frag_load(Qb + 32, HD);

  v8f oacc[4] = {v8f{}, v8f{}, v8f{}, v8f{}};
  float mrun[8], lrun[8];
#pragma unroll
  for (int r = 0; r < 8; ++r) { mrun[r] = -1e30f; lrun[r] = 0.f; }

  wait_async0();
  wg_barrier();

  for (int n0 = 0; n0 < NSEQ; n0 += 32) {
    const int cur = (n0 >> 5) & 1;
    if (n0 + 32 < NSEQ) STAGE_KV(cur ^ 1, n0 + 32);
    const f16* Kc = &Kl[cur][0];
    const f16* Vc = &Vl[cur][0];

    // S tile 16x32 = Q(16x64) @ K^T(64x32), K fragments from LDS
    v8f s0 = {}, s1 = {};
    s0 = wmma_f16(qa0, frag_load(Kc + 0 * 64 + 0, 64), s0);
    s0 = wmma_f16(qa1, frag_load(Kc + 0 * 64 + 32, 64), s0);
    s1 = wmma_f16(qa0, frag_load(Kc + 16 * 64 + 0, 64), s1);
    s1 = wmma_f16(qa1, frag_load(Kc + 16 * 64 + 32, 64), s1);

    // online softmax; rows live in 16-lane half-groups -> xor masks 1,2,4,8
#pragma unroll
    for (int r = 0; r < 8; ++r) {
      float bm = fmaxf(s0[r], s1[r]);
      bm = fmaxf(bm, __shfl_xor(bm, 1)); bm = fmaxf(bm, __shfl_xor(bm, 2));
      bm = fmaxf(bm, __shfl_xor(bm, 4)); bm = fmaxf(bm, __shfl_xor(bm, 8));
      const float mn = fmaxf(mrun[r], bm);
      const float alpha = __expf(mrun[r] - mn);
      mrun[r] = mn;
      const float p0 = __expf(s0[r] - mn);
      const float p1 = __expf(s1[r] - mn);
      float rs = p0 + p1;
      rs += __shfl_xor(rs, 1); rs += __shfl_xor(rs, 2);
      rs += __shfl_xor(rs, 4); rs += __shfl_xor(rs, 8);
      lrun[r] = lrun[r] * alpha + rs;
#pragma unroll
      for (int dc = 0; dc < 4; ++dc) oacc[dc][r] *= alpha;
      const int m = r + 8 * grp;
      pt[m * 32 + lr]      = (f16)p0;
      pt[m * 32 + 16 + lr] = (f16)p1;
    }

    // O += P(16x32) @ V(32x64); V fragments from LDS (V^T rows contiguous)
    const v16h pa = frag_load(pt, 32);
#pragma unroll
    for (int dc = 0; dc < 4; ++dc)
      oacc[dc] = wmma_f16(pa, frag_load(Vc + 16 * dc * 32, 32), oacc[dc]);

    wait_async0();
    wg_barrier();
  }
#undef STAGE_KV

  const int b = bh / HEADS, h = bh - b * HEADS;
#pragma unroll
  for (int r = 0; r < 8; ++r) {
    const float inv = 1.0f / lrun[r];
    const size_t row = (size_t)b * NSEQ + m0 + r + 8 * grp;
#pragma unroll
    for (int dc = 0; dc < 4; ++dc)
      O[row * DIMC + h * HD + dc * 16 + lr] = (f16)(oacc[dc][r] * inv);
  }
}

// ------------------------------------------------------------- launcher
extern "C" void kernel_launch(void* const* d_in, const int* in_sizes, int n_in,
                              void* d_out, int out_size, void* d_ws, size_t ws_size,
                              hipStream_t stream) {
  (void)in_sizes; (void)n_in; (void)out_size; (void)ws_size;
  const float* x      = (const float*)d_in[0];
  const float* qkv_w  = (const float*)d_in[1];
  const float* proj_w = (const float*)d_in[2];
  const float* proj_b = (const float*)d_in[3];
  const float* ln1_w  = (const float*)d_in[4];
  const float* ln1_b  = (const float*)d_in[5];
  const float* ln2_w  = (const float*)d_in[6];
  const float* ln2_b  = (const float*)d_in[7];
  const float* fc1_w  = (const float*)d_in[8];
  const float* fc1_b  = (const float*)d_in[9];
  const float* fc2_w  = (const float*)d_in[10];
  const float* fc2_b  = (const float*)d_in[11];
  float* out = (float*)d_out;

  char* ws = (char*)d_ws;
  size_t off = 0;
  auto alloc_h = [&](size_t elems) {
    f16* p = (f16*)(ws + off);
    off += (elems * sizeof(f16) + 255) & ~(size_t)255;
    return p;
  };
  auto alloc_f = [&](size_t elems) {
    float* p = (float*)(ws + off);
    off += (elems * sizeof(float) + 255) & ~(size_t)255;
    return p;
  };

  f16* wQKV = alloc_h((size_t)3 * DIMC * DIMC);
  f16* wPRJ = alloc_h((size_t)DIMC * DIMC);
  f16* wFC1 = alloc_h((size_t)HIDDEN * DIMC);
  f16* wFC2 = alloc_h((size_t)DIMC * HIDDEN);
  f16* y1   = alloc_h((size_t)NTOK * DIMC);
  f16* qd   = alloc_h((size_t)2 * HEADS * NSEQ * HD);
  f16* kd   = alloc_h((size_t)2 * HEADS * NSEQ * HD);
  f16* vtd  = alloc_h((size_t)2 * HEADS * NSEQ * HD);
  f16* aO   = alloc_h((size_t)NTOK * DIMC);
  float* pO = alloc_f((size_t)NTOK * DIMC);
  f16* y2   = alloc_h((size_t)NTOK * DIMC);
  f16* hb   = alloc_h((size_t)NTOK * HIDDEN);

  auto cvt = [&](const float* src, f16* dst, int n) {
    cvt_kernel<<<(n + 255) / 256, 256, 0, stream>>>(src, dst, n);
  };
  cvt(qkv_w, wQKV, 3 * DIMC * DIMC);
  cvt(proj_w, wPRJ, DIMC * DIMC);
  cvt(fc1_w, wFC1, HIDDEN * DIMC);
  cvt(fc2_w, wFC2, DIMC * HIDDEN);

  ln1_kernel<<<NTOK / 8, 256, 0, stream>>>(x, ln1_w, ln1_b, y1);

  qkv_gemm_kernel<<<dim3(3 * DIMC / 64, NTOK / 256), 256, 0, stream>>>(
      y1, wQKV, qd, kd, vtd);

  attn_kernel<<<2 * HEADS * (NSEQ / 128), 256, 0, stream>>>(qd, kd, vtd, aO);

  gemm_kernel<EPI_BIAS_F32, true><<<dim3(DIMC / 64, NTOK / 256), 256, 0, stream>>>(
      aO, wPRJ, proj_b, pO, DIMC, DIMC);

  ln2_kernel<<<NTOK / 8, 256, 0, stream>>>(x, pO, ln2_w, ln2_b, y2);

  gemm_kernel<EPI_GELU_F16, true><<<dim3(HIDDEN / 64, NTOK / 256), 256, 0, stream>>>(
      y2, wFC1, fc1_b, hb, HIDDEN, DIMC);

  gemm_kernel<EPI_BIAS2X_F32, false><<<dim3(DIMC / 64, NTOK / 256), 256, 0, stream>>>(
      hb, wFC2, fc2_b, out, DIMC, HIDDEN);
}
